// ProvHyperGNN_67577015435767
// MI455X (gfx1250) — matrix operand order repaired and checked by reading, not compiled, not run on hardware
//
#include <hip/hip_runtime.h>
#include <hip/hip_bf16.h>

// CDNA5 / gfx1250 implementation of the ProvHyperGNN reference.
// Wave32, WMMA f16 GEMM (weights resident in VGPRs, persistent waves) fused
// with hyperedge scatter; gather+ReLU+pool fused with LDS-staged reduction.
// All large-buffer indexing is 32-bit so global ops use SGPR-base + 32-bit
// VGPR-offset (GVS) addressing instead of 64-bit per-lane address math.

typedef __attribute__((ext_vector_type(16))) _Float16 v16h;
typedef __attribute__((ext_vector_type(8)))  float    v8f;

#define N_NODES 50000
#define N_EDGES 500000
#define HID     64
#define NGRAPH  64

// ---------------- preprocessing kernels ----------------

__global__ void k_deg(const int* __restrict__ ei, int* __restrict__ deg, int n2e) {
  int t = blockIdx.x * blockDim.x + threadIdx.x;
  if (t < n2e) atomicAdd(&deg[ei[t]], 1);
}

__global__ void k_binv(const int* __restrict__ deg, float* __restrict__ binv, int n) {
  int v = blockIdx.x * blockDim.x + threadIdx.x;
  if (v < n) {
    int d = deg[v];
    binv[v] = (d > 1) ? 1.0f / (float)d : 0.0f;   // deg==1 incidences dropped -> B=0; deg==0 -> 0
  }
}

__global__ void k_meta(const int* __restrict__ ei, const int* __restrict__ deg,
                       const float* __restrict__ nimp, const int* __restrict__ batch,
                       float* __restrict__ Dinv, float* __restrict__ imp,
                       int* __restrict__ ebat, int E) {
  int e = blockIdx.x * blockDim.x + threadIdx.x;
  if (e < E) {
    int u = ei[e], w = ei[E + e];
    int ku = (deg[u] != 1) ? 1 : 0;
    int kw = (deg[w] != 1) ? 1 : 0;
    Dinv[e] = 1.0f / (float)(1 + ku + kw);        // kept incidences + self loop
    imp[e]  = fminf(nimp[u], nimp[w]);
    ebat[e] = batch[u];
  }
}

// ---------------- fused WMMA GEMM + hyperedge scatter ----------------
// Persistent waves: each wave keeps all B (weight) fragments in VGPRs for the
// whole kernel and grid-strides over 16-row tiles of the E x DIN activation.
// A lane layout (16-bit A 16x32): lanes 0-15 K=0..15, lanes 16-31
// K=16..31 (contiguous K order used here). B lane layout (32x16): lanes 0-15 K=0..15, lanes 16-31
// K=16..31; col = lane&15. C: VGPR r -> row r / r+8 by lane half, col=lane&15.

template <int DIN, bool SCALE>
__global__ void __launch_bounds__(128)
k_gemm_scatter(const float* __restrict__ src,
               const float* __restrict__ rowScale,           // layer0: edge_imp
               const float* __restrict__ W,                  // DIN x 64 row-major f32
               const int* __restrict__ ei,
               float* __restrict__ xw, float* __restrict__ he, int E) {
  constexpr int KC = DIN / 32;
  const int lane = threadIdx.x & 31;
  const int wave = threadIdx.x >> 5;
  const int ntiles = E >> 4;
  const int wstride = gridDim.x * 4;

  const int kh    = (lane >> 4) * 16;   // A K sub-offset for this lane half
  const int koff  = (lane >> 4) * 16;   // B K sub-offset for this lane half
  const int cb    = lane & 15;          // B/C column within 16-wide tile
  const int rbase = (lane >> 4) * 8;    // C row base for this lane half

  // Weight fragments: loaded/converted once, resident in VGPRs for all tiles.
  v16h Bf[KC][4];
#pragma unroll
  for (int k = 0; k < KC; ++k) {
    const float* Wc = W + (k * 32 + koff) * HID;
#pragma unroll
    for (int i = 0; i < 16; ++i) {
      const float* wr = Wc + i * HID;
      Bf[k][0][i] = (_Float16)wr[cb];
      Bf[k][1][i] = (_Float16)wr[16 + cb];
      Bf[k][2][i] = (_Float16)wr[32 + cb];
      Bf[k][3][i] = (_Float16)wr[48 + cb];
    }
  }

  for (int tile = blockIdx.x * 4 + wave; tile < ntiles; tile += wstride) {
    const int m0  = tile * 16;
    const int row = m0 + (lane & 15);

    // Prefetch next strip's A rows (global_prefetch_b8) to hide the only
    // serial per-tile dependency (A load -> cvt -> wmma).
    const int nt = tile + wstride;
    if (nt < ntiles)
      __builtin_prefetch(src + (nt * 16 + (lane & 15)) * DIN, 0, 1);

    float s = 1.0f;
    if (SCALE) s = rowScale[row];

    v8f acc0 = {}, acc1 = {}, acc2 = {}, acc3 = {};
#pragma unroll
    for (int k = 0; k < KC; ++k) {
      const float* hp = src + row * DIN + k * 32;     // 32-bit index math
      float4 a0 = *(const float4*)(hp + kh);
      float4 a1 = *(const float4*)(hp + kh + 4);
      float4 a2 = *(const float4*)(hp + kh + 8);
      float4 a3 = *(const float4*)(hp + kh + 12);

      v16h A;
      A[0]  = (_Float16)(s * a0.x); A[1]  = (_Float16)(s * a0.y);
      A[2]  = (_Float16)(s * a0.z); A[3]  = (_Float16)(s * a0.w);
      A[4]  = (_Float16)(s * a1.x); A[5]  = (_Float16)(s * a1.y);
      A[6]  = (_Float16)(s * a1.z); A[7]  = (_Float16)(s * a1.w);
      A[8]  = (_Float16)(s * a2.x); A[9]  = (_Float16)(s * a2.y);
      A[10] = (_Float16)(s * a2.z); A[11] = (_Float16)(s * a2.w);
      A[12] = (_Float16)(s * a3.x); A[13] = (_Float16)(s * a3.y);
      A[14] = (_Float16)(s * a3.z); A[15] = (_Float16)(s * a3.w);

      acc0 = __builtin_amdgcn_wmma_f32_16x16x32_f16(false, A, false, Bf[k][0], (short)0, acc0, false, false);
      acc1 = __builtin_amdgcn_wmma_f32_16x16x32_f16(false, A, false, Bf[k][1], (short)0, acc1, false, false);
      acc2 = __builtin_amdgcn_wmma_f32_16x16x32_f16(false, A, false, Bf[k][2], (short)0, acc2, false, false);
      acc3 = __builtin_amdgcn_wmma_f32_16x16x32_f16(false, A, false, Bf[k][3], (short)0, acc3, false, false);
    }

#pragma unroll
    for (int r = 0; r < 8; ++r) {
      const int m = m0 + rbase + r;
      const int u = ei[m];
      const int w = ei[E + m];
      const float v0 = acc0[r], v1 = acc1[r], v2 = acc2[r], v3 = acc3[r];
      const int mb = m * HID + cb;                    // <= 32M elements: int-safe
      xw[mb]      = v0;
      xw[mb + 16] = v1;
      xw[mb + 32] = v2;
      xw[mb + 48] = v3;
      const int ub = u * HID + cb, wb = w * HID + cb; // <= 3.2M elements
      atomicAdd(&he[ub],      v0); atomicAdd(&he[ub + 16], v1);
      atomicAdd(&he[ub + 32], v2); atomicAdd(&he[ub + 48], v3);
      atomicAdd(&he[wb],      v0); atomicAdd(&he[wb + 16], v1);
      atomicAdd(&he[wb + 32], v2); atomicAdd(&he[wb + 48], v3);
    }
  }
}

// ---------------- gather + bias + ReLU + imp + LDS-staged pooling ----------------
// 256 threads / block, 64 edges / block, 16 cols per thread (4x float4).

__global__ void __launch_bounds__(256)
k_gather(const float* __restrict__ xw, const float* __restrict__ he,
         const int* __restrict__ ei, const float* __restrict__ binv,
         const float* __restrict__ Dinv, const float* __restrict__ imp,
         const int* __restrict__ ebat, const float* __restrict__ bias,
         float* __restrict__ hnext, float* __restrict__ pool, int layer, int E) {
  __shared__ float lpool[NGRAPH * HID];                  // 16 KB of the 320 KB WGP LDS
  for (int i = threadIdx.x; i < NGRAPH * HID; i += 256) lpool[i] = 0.0f;
  __syncthreads();

  const int e = blockIdx.x * 64 + (threadIdx.x >> 2);
  if (e < E) {
    const int u = ei[e], w = ei[E + e];
    const float bu = binv[u], bw = binv[w];
    const float d  = Dinv[e], im = imp[e];
    const int   g  = ebat[e];
    const int   j0 = (threadIdx.x & 3) * 16;
    const int eb = e * HID;                            // 32-bit index math
    const int ub = u * HID, wb = w * HID;
#pragma unroll
    for (int jj = 0; jj < 16; jj += 4) {
      const int j = j0 + jj;
      float4 hu = *(const float4*)(he + ub + j);
      float4 hw = *(const float4*)(he + wb + j);
      float4 xv = *(const float4*)(xw + eb + j);
      float4 bb = *(const float4*)(bias + j);
      float4 o;
      o.x = fmaxf(d * (bu * hu.x + bw * hw.x + xv.x) + bb.x, 0.0f) * im;
      o.y = fmaxf(d * (bu * hu.y + bw * hw.y + xv.y) + bb.y, 0.0f) * im;
      o.z = fmaxf(d * (bu * hu.z + bw * hw.z + xv.z) + bb.z, 0.0f) * im;
      o.w = fmaxf(d * (bu * hu.w + bw * hw.w + xv.w) + bb.w, 0.0f) * im;
      *(float4*)(hnext + eb + j) = o;
      atomicAdd(&lpool[g * HID + j + 0], o.x);           // ds_add_f32
      atomicAdd(&lpool[g * HID + j + 1], o.y);
      atomicAdd(&lpool[g * HID + j + 2], o.z);
      atomicAdd(&lpool[g * HID + j + 3], o.w);
    }
  }
  __syncthreads();
  for (int i = threadIdx.x; i < NGRAPH * HID; i += 256) {
    float v = lpool[i];
    if (v != 0.0f)
      atomicAdd(&pool[(i >> 6) * (3 * HID) + layer * HID + (i & 63)], v);
  }
}

// ---------------- host-side launch ----------------

extern "C" void kernel_launch(void* const* d_in, const int* in_sizes, int n_in,
                              void* d_out, int out_size, void* d_ws, size_t ws_size,
                              hipStream_t stream) {
  (void)in_sizes; (void)n_in; (void)ws_size;
  // setup_inputs order: x, edge_attr, edge_index, batch, node_imp, W0,b0, W1,b1, W2,b2, num_graphs
  const float* edge_attr = (const float*)d_in[1];
  const int*   ei        = (const int*)d_in[2];   // (2,E) int32 (JAX default x64-off)
  const int*   batch     = (const int*)d_in[3];
  const float* nimp      = (const float*)d_in[4];
  const float* Ws[3] = {(const float*)d_in[5], (const float*)d_in[7], (const float*)d_in[9]};
  const float* bs[3] = {(const float*)d_in[6], (const float*)d_in[8], (const float*)d_in[10]};
  float* out = (float*)d_out;

  char* wsp = (char*)d_ws;
  auto alloc = [&](size_t bytes) {
    char* p = wsp;
    wsp += (bytes + 255) & ~(size_t)255;
    return p;
  };
  int*   deg  = (int*)  alloc((size_t)N_NODES * 4);
  float* binv = (float*)alloc((size_t)N_NODES * 4);
  float* Dinv = (float*)alloc((size_t)N_EDGES * 4);
  float* imp  = (float*)alloc((size_t)N_EDGES * 4);
  int*   ebat = (int*)  alloc((size_t)N_EDGES * 4);
  float* he   = (float*)alloc((size_t)N_NODES * HID * 4);
  float* bufA = (float*)alloc((size_t)N_EDGES * HID * 4);   // h (gather output)
  float* bufB = (float*)alloc((size_t)N_EDGES * HID * 4);   // xw (gemm output)

  hipMemsetAsync(deg, 0, (size_t)N_NODES * 4, stream);
  hipMemsetAsync(d_out, 0, (size_t)out_size * sizeof(float), stream);

  k_deg <<<(2 * N_EDGES + 255) / 256, 256, 0, stream>>>(ei, deg, 2 * N_EDGES);
  k_binv<<<(N_NODES + 255) / 256,     256, 0, stream>>>(deg, binv, N_NODES);
  k_meta<<<(N_EDGES + 255) / 256,     256, 0, stream>>>(ei, deg, nimp, batch, Dinv, imp, ebat, N_EDGES);

  const int gemm_blocks = 1024;                 // ~8 tiles per persistent wave
  for (int l = 0; l < 3; ++l) {
    hipMemsetAsync(he, 0, (size_t)N_NODES * HID * 4, stream);
    if (l == 0) {
      k_gemm_scatter<32, true><<<gemm_blocks, 128, 0, stream>>>(
          edge_attr, imp, Ws[0], ei, bufB, he, N_EDGES);
    } else {
      k_gemm_scatter<64, false><<<gemm_blocks, 128, 0, stream>>>(
          bufA, nullptr, Ws[l], ei, bufB, he, N_EDGES);
    }
    k_gather<<<(N_EDGES + 63) / 64, 256, 0, stream>>>(bufB, he, ei, binv, Dinv, imp, ebat,
                                                      bs[l], bufA, out, l, N_EDGES);
  }
}
